// SignLinear_43696997270100
// MI455X (gfx1250) — matrix operand order repaired and checked by reading, not compiled, not run on hardware
//
#include <hip/hip_runtime.h>
#include <math.h>

// ---------------------------------------------------------------------------
// SignLinear fused forward for MI455X (gfx1250, wave32, WMMA).
//   mu   = p_x @ c^T + sum(d)
//   ssq  = p_x^2 @ (c^2)^T + 2 * p_x @ (c*d)^T + sum(d^2)
//   out  = 0.5 - 0.5*erf((2047.5 - mu) / (sqrt(mu - ssq + eps) * sqrt(2)))
// Three f16 GEMMs with shared operands, f32 accumulation via
// V_WMMA_F32_16X16X32_F16; double-buffered LDS staging using
// GLOBAL_LOAD_ASYNC_TO_LDS_B128 (ASYNCcnt) when the toolchain exposes it.
// ---------------------------------------------------------------------------

typedef _Float16 h4   __attribute__((ext_vector_type(4)));
typedef _Float16 h8   __attribute__((ext_vector_type(8)));
typedef _Float16 v16h __attribute__((ext_vector_type(16)));
typedef float    v8f  __attribute__((ext_vector_type(8)));
typedef int      v4i  __attribute__((ext_vector_type(4)));

#define N_ROWS 8192
#define IN_F   4096
#define OUT_F  4096

#define BM 64
#define BN 128
#define KC 32
#define NSTEPS (IN_F / KC)

#if defined(__gfx1250__) && __has_builtin(__builtin_amdgcn_global_load_async_to_lds_b128)
#define HAVE_ASYNC_LDS 1
#else
#define HAVE_ASYNC_LDS 0
#endif

union Frag16 { v16h v; h8 h[2]; };

// 16-byte global -> LDS copy: async DMA path on gfx1250, sync fallback.
// Probe-confirmed: builtin has 4 args; arg0 is (AS1) int4*, arg1 the LDS side.
__device__ __forceinline__ void cp16(const _Float16* g, _Float16* l) {
#if HAVE_ASYNC_LDS
    __builtin_amdgcn_global_load_async_to_lds_b128(
        (__attribute__((address_space(1))) v4i*)g,
        (__attribute__((address_space(3))) v4i*)l,
        0, 0);
#else
    *(h8*)l = *(const h8*)g;
#endif
}

#if HAVE_ASYNC_LDS
template <int N>
__device__ __forceinline__ void wait_asynccnt() {
#if __has_builtin(__builtin_amdgcn_s_wait_asynccnt)
    __builtin_amdgcn_s_wait_asynccnt(N);
#else
    asm volatile("s_wait_asynccnt %0" ::"n"(N));
#endif
}
#endif

// ---------------------------------------------------------------------------
// Kernel 1: per-output-row weight transform.
// c = 2*sigmoid(w)-1, d = 1-sigmoid(w); emit f16 {c, c^2, c*d} and row sums.
// ---------------------------------------------------------------------------
__global__ __launch_bounds__(256)
void prep_weight_kernel(const float* __restrict__ w,
                        _Float16* __restrict__ c16,
                        _Float16* __restrict__ c2_16,
                        _Float16* __restrict__ cd16,
                        float* __restrict__ sum_d,
                        float* __restrict__ sum_d2) {
    const int o = blockIdx.x;
    const int t = threadIdx.x;
    const size_t base = (size_t)o * IN_F;
    float s1 = 0.0f, s2 = 0.0f;
    for (int i = t; i < IN_F; i += 256) {
        float wv = w[base + i];
        float pw = 1.0f / (1.0f + __expf(-wv));
        float c  = 2.0f * pw - 1.0f;
        float d  = 1.0f - pw;
        c16 [base + i] = (_Float16)c;
        c2_16[base + i] = (_Float16)(c * c);
        cd16[base + i] = (_Float16)(c * d);
        s1 += d;
        s2 += d * d;
    }
    __shared__ float r1[256];
    __shared__ float r2[256];
    r1[t] = s1; r2[t] = s2;
    __syncthreads();
    #pragma unroll
    for (int s = 128; s > 0; s >>= 1) {
        if (t < s) { r1[t] += r1[t + s]; r2[t] += r2[t + s]; }
        __syncthreads();
    }
    if (t == 0) { sum_d[o] = r1[0]; sum_d2[o] = r2[0]; }
}

// ---------------------------------------------------------------------------
// Kernel 2: p_x (f32) -> f16 p_x and f16 p_x^2 (4 elements per thread).
// ---------------------------------------------------------------------------
__global__ __launch_bounds__(256)
void prep_px_kernel(const float* __restrict__ px,
                    _Float16* __restrict__ px16,
                    _Float16* __restrict__ px2_16) {
    size_t idx = ((size_t)blockIdx.x * blockDim.x + threadIdx.x) * 4;
    float4 v = *(const float4*)(px + idx);
    h4 a = { (_Float16)v.x, (_Float16)v.y, (_Float16)v.z, (_Float16)v.w };
    h4 b = { (_Float16)(v.x * v.x), (_Float16)(v.y * v.y),
             (_Float16)(v.z * v.z), (_Float16)(v.w * v.w) };
    *(h4*)(px16  + idx) = a;
    *(h4*)(px2_16 + idx) = b;
}

// ---------------------------------------------------------------------------
// Kernel 3: fused triple-GEMM + epilogue.
// Block tile 64(M) x 128(N), 8 waves; each wave owns 32x32 outputs
// (2x2 WMMA tiles x 3 accumulators = 12 wmma per 32-deep K step).
// Double-buffered LDS pipeline: stage k+1 issued (async) before computing k.
// ---------------------------------------------------------------------------
__global__ __launch_bounds__(256)
void signlinear_gemm_kernel(const _Float16* __restrict__ px16,
                            const _Float16* __restrict__ px2_16,
                            const _Float16* __restrict__ c16,
                            const _Float16* __restrict__ c2_16,
                            const _Float16* __restrict__ cd16,
                            const float* __restrict__ sum_d,
                            const float* __restrict__ sum_d2,
                            float* __restrict__ out) {
    __shared__ _Float16 sA1[2][BM * KC];   // p_x tiles
    __shared__ _Float16 sA2[2][BM * KC];   // p_x^2 tiles
    __shared__ _Float16 sB1[2][BN * KC];   // c tiles      (n-major rows, contiguous K)
    __shared__ _Float16 sB2[2][BN * KC];   // c^2 tiles
    __shared__ _Float16 sB3[2][BN * KC];   // c*d tiles

    const int tid  = threadIdx.x;
    const int lane = tid & 31;
    const int wave = tid >> 5;
    const int wm   = wave & 1;   // 2 wave rows  -> 32 M each
    const int wn   = wave >> 1;  // 4 wave cols  -> 32 N each
    const int m_blk = blockIdx.y * BM;
    const int n_blk = blockIdx.x * BN;

    const int half = lane >> 4;  // 0/1: lane halves per CDNA5 WMMA layouts
    const int l16  = lane & 15;

    v8f acc1[2][2] = {};  // p_x   @ c^T
    v8f acc2[2][2] = {};  // p_x^2 @ (c^2)^T
    v8f acc3[2][2] = {};  // p_x   @ (c*d)^T

    const int arow   = tid >> 2;        // 0..63
    const int achunk = (tid & 3) * 8;   // 0,8,16,24

    // Issue one full stage (8 x 16B per thread) into LDS buffer `buf`.
    auto stage = [&](int k0, int buf) {
        size_t ga = (size_t)(m_blk + arow) * IN_F + (size_t)(k0 + achunk);
        cp16(&px16[ga],   &sA1[buf][arow * KC + achunk]);
        cp16(&px2_16[ga], &sA2[buf][arow * KC + achunk]);
        #pragma unroll
        for (int p = 0; p < 2; ++p) {
            int idx    = tid + p * 256;
            int brow   = idx >> 2;          // 0..127
            int bchunk = (idx & 3) * 8;
            size_t gb = (size_t)(n_blk + brow) * IN_F + (size_t)(k0 + bchunk);
            cp16(&c16[gb],   &sB1[buf][brow * KC + bchunk]);
            cp16(&c2_16[gb], &sB2[buf][brow * KC + bchunk]);
            cp16(&cd16[gb],  &sB3[buf][brow * KC + bchunk]);
        }
    };

    stage(0, 0);  // prologue: stage 0 in flight (8 async ops)

    for (int step = 0; step < NSTEPS; ++step) {
        const int cur = step & 1;
        // Prefetch next stage into the other buffer (safe: all waves passed
        // the trailing barrier of the previous iteration before re-writing it).
        if (step + 1 < NSTEPS) stage((step + 1) * KC, cur ^ 1);
#if HAVE_ASYNC_LDS
        // Async loads retire in order per wave: waiting to <=8 outstanding
        // guarantees the 8 copies of stage `step` have landed in LDS.
        if (step + 1 < NSTEPS) wait_asynccnt<8>();
        else                   wait_asynccnt<0>();
#endif
        __syncthreads();

        // --- A fragments: 16x32 MxK; lane holds row l16, contiguous K (k = 16*half + s), same order as B
        Frag16 a1[2], a2[2];
        #pragma unroll
        for (int mt = 0; mt < 2; ++mt) {
            const int row = wm * 32 + mt * 16 + l16;
            const _Float16* ap = &sA1[cur][row * KC];
            a1[mt].h[0] = *(const h8*)(ap + half * 16);
            a1[mt].h[1] = *(const h8*)(ap + half * 16 + 8);
            const _Float16* aq = &sA2[cur][row * KC];
            a2[mt].h[0] = *(const h8*)(aq + half * 16);
            a2[mt].h[1] = *(const h8*)(aq + half * 16 + 8);
        }

        // --- B fragments + 12 WMMAs
        #pragma unroll
        for (int nt = 0; nt < 2; ++nt) {
            const int col = wn * 32 + nt * 16 + l16;
            Frag16 b1, b2, b3;
            const _Float16* bp1 = &sB1[cur][col * KC + half * 16];
            b1.h[0] = *(const h8*)(bp1);
            b1.h[1] = *(const h8*)(bp1 + 8);
            const _Float16* bp2 = &sB2[cur][col * KC + half * 16];
            b2.h[0] = *(const h8*)(bp2);
            b2.h[1] = *(const h8*)(bp2 + 8);
            const _Float16* bp3 = &sB3[cur][col * KC + half * 16];
            b3.h[0] = *(const h8*)(bp3);
            b3.h[1] = *(const h8*)(bp3 + 8);
            #pragma unroll
            for (int mt = 0; mt < 2; ++mt) {
                acc1[mt][nt] = __builtin_amdgcn_wmma_f32_16x16x32_f16(
                    false, a1[mt].v, false, b1.v, (short)0, acc1[mt][nt], false, false);
                acc2[mt][nt] = __builtin_amdgcn_wmma_f32_16x16x32_f16(
                    false, a2[mt].v, false, b2.v, (short)0, acc2[mt][nt], false, false);
                acc3[mt][nt] = __builtin_amdgcn_wmma_f32_16x16x32_f16(
                    false, a1[mt].v, false, b3.v, (short)0, acc3[mt][nt], false, false);
            }
        }
        __syncthreads();
    }

    // --- epilogue: C/D layout VGPR r -> M = r + 8*half, N = l16
    const float inv_root2 = 0.70710678118654752440f;
    #pragma unroll
    for (int nt = 0; nt < 2; ++nt) {
        const int gn = n_blk + wn * 32 + nt * 16 + l16;
        const float sd  = sum_d[gn];
        const float sd2 = sum_d2[gn];
        #pragma unroll
        for (int mt = 0; mt < 2; ++mt) {
            const int m_base = m_blk + wm * 32 + mt * 16 + half * 8;
            #pragma unroll
            for (int r = 0; r < 8; ++r) {
                float mu  = acc1[mt][nt][r] + sd;
                float ssq = acc2[mt][nt][r] + 2.0f * acc3[mt][nt][r] + sd2;
                float var = mu - ssq + 1e-6f;
                float sigma = sqrtf(var);
                // threshold - 0.5 = ceil(4096/2) - 0.5 = 2047.5
                float z = (2047.5f - mu) / sigma * inv_root2;
                out[(size_t)(m_base + r) * OUT_F + gn] = 0.5f - 0.5f * erff(z);
            }
        }
    }
}

// ---------------------------------------------------------------------------
extern "C" void kernel_launch(void* const* d_in, const int* in_sizes, int n_in,
                              void* d_out, int out_size, void* d_ws, size_t ws_size,
                              hipStream_t stream) {
    const float* p_x    = (const float*)d_in[0];   // (8192, 4096) f32
    const float* weight = (const float*)d_in[1];   // (4096, 4096) f32
    float* out = (float*)d_out;                    // (8192, 4096) f32

    char* ws = (char*)d_ws;
    size_t off = 0;
    _Float16* c16    = (_Float16*)(ws + off); off += (size_t)OUT_F * IN_F * sizeof(_Float16);
    _Float16* c2_16  = (_Float16*)(ws + off); off += (size_t)OUT_F * IN_F * sizeof(_Float16);
    _Float16* cd16   = (_Float16*)(ws + off); off += (size_t)OUT_F * IN_F * sizeof(_Float16);
    _Float16* px16   = (_Float16*)(ws + off); off += (size_t)N_ROWS * IN_F * sizeof(_Float16);
    _Float16* px2_16 = (_Float16*)(ws + off); off += (size_t)N_ROWS * IN_F * sizeof(_Float16);
    float*    sum_d  = (float*)(ws + off);    off += (size_t)OUT_F * sizeof(float);
    float*    sum_d2 = (float*)(ws + off);    off += (size_t)OUT_F * sizeof(float);

    prep_weight_kernel<<<OUT_F, 256, 0, stream>>>(weight, c16, c2_16, cd16, sum_d, sum_d2);

    const int px_blocks = (int)(((size_t)N_ROWS * IN_F) / (256 * 4));
    prep_px_kernel<<<px_blocks, 256, 0, stream>>>(p_x, px16, px2_16);

    dim3 grid(OUT_F / BN, N_ROWS / BM);
    signlinear_gemm_kernel<<<grid, 256, 0, stream>>>(
        px16, px2_16, c16, c2_16, cd16, sum_d, sum_d2, out);
}
